// TRFlow_occjoint_53480932770043
// MI455X (gfx1250) — matrix operand order repaired and checked by reading, not compiled, not run on hardware
//
#include <hip/hip_runtime.h>

// ===== CDNA5 (gfx1250) WMMA / vector types =====
typedef __attribute__((ext_vector_type(16))) _Float16     v16h;
typedef __attribute__((ext_vector_type(8)))  float        v8f;
typedef __attribute__((ext_vector_type(8)))  unsigned int v8u;
typedef __attribute__((ext_vector_type(4)))  unsigned int u32x4;
typedef __attribute__((ext_vector_type(8)))  int          i32x8;
typedef __attribute__((ext_vector_type(4)))  int          i32x4;

#define TB    256            // 8 wave32 per workgroup
#define DIVF  0.05f
#define APW   20             // A LDS row pitch in uint32 (32 f16 + pad)
#define BPW   68             // B LDS row pitch in uint32 (64 + 4 pad dwords, via TDM pad)
#define ABYTES (128 * APW * 4)               // one A buffer: 128 rows
#define BBYTES (16 * BPW * 4)                // one B buffer: 16 k-pair rows
#define SMEMB  (2 * ABYTES + 2 * BBYTES)     // double-buffered ping-pong

union F16x2 { unsigned int u; _Float16 h[2]; };

__device__ __forceinline__ float leakyf(float v) { return v >= 0.f ? v : 0.1f * v; }

// ---------------------------------------------------------------------------
// Weight pre-pack: OIHW f32 -> [tap][ci_pair][co] packed-f16-pair (uint32),
// zero-padded so each 32-input-channel K chunk stays inside its tap rows.
// This is exactly the WMMA B-fragment global layout, enabling the TDM to DMA
// weight tiles straight into LDS with no per-thread format shuffling.
// ---------------------------------------------------------------------------
__global__ void pack_w_k(const float* __restrict__ w, unsigned* __restrict__ out,
                         int Cin, int Cout, int KPp, long n) {
  long i = (long)blockIdx.x * blockDim.x + threadIdx.x;
  if (i >= n) return;
  int co = (int)(i % Cout); long t = i / Cout;
  int kp = (int)(t % KPp);  int tap = (int)(t / KPp);
  int ky = tap / 3, kx = tap - 3 * ky;
  int c0 = 2 * kp;
  F16x2 p;
  p.h[0] = (_Float16)((c0     < Cin) ? w[(((long)co * Cin + c0    ) * 3 + ky) * 3 + kx] : 0.f);
  p.h[1] = (_Float16)((c0 + 1 < Cin) ? w[(((long)co * Cin + c0 + 1) * 3 + ky) * 3 + kx] : 0.f);
  out[i] = p.u;
}

// ---------------------------------------------------------------------------
// Implicit-GEMM 3x3 conv:  M = Ho*Wo, N = Cout, K = Cin*9.
//   Workgroup tile 128(M) x 64(N); each wave: 32M x 32N = 4 WMMAs per K-step.
//   Double-buffered pipeline: per K-step, issue the TDM weight-tile DMA for
//   step i+1 right after the barrier, run the 4 WMMAs of step i from the other
//   buffer, stage the A tile for i+1 (branch-free f32->packed-f16; all H/W are
//   powers of two -> shift/mask), s_wait_tensorcnt, one barrier. The TDM DMA
//   overlaps WMMA + A staging; one s_barrier per K-step.
// ---------------------------------------------------------------------------
__global__ __launch_bounds__(TB) void conv3x3_wmma(
    const float* __restrict__ x, int Cx, int ci0, int Cin, int H, int W,
    const unsigned* __restrict__ wpk, int KPp, const float* __restrict__ bias,
    float* __restrict__ y, int Cy, int co0, int Cout, int Ho, int Wo, int lwo,
    int stride, int dil, int relu)
{
  extern __shared__ char smem[];
  unsigned* As32 = (unsigned*)smem;                  // [2][128][APW]
  unsigned* Bs32 = (unsigned*)(smem + 2 * ABYTES);   // [2][16][BPW]

  const int tid  = threadIdx.x;
  const int lane = tid & 31;
  const int wid  = tid >> 5;
  const int wm   = wid & 3;        // wave M index (0..3) -> 32 rows each
  const int wn   = wid >> 2;       // wave N index (0..1) -> 32 cols each
  const int hl   = lane >> 4;
  const int l16  = lane & 15;

  const int M   = Ho * Wo;
  const int M0  = blockIdx.x * 128;
  const int N0  = blockIdx.y * 64;
  const int b   = blockIdx.z;
  const int Wm1 = Wo - 1;

  v8f acc[2][2];
#pragma unroll
  for (int i = 0; i < 8; ++i) {
    acc[0][0][i] = 0.f; acc[0][1][i] = 0.f; acc[1][0][i] = 0.f; acc[1][1][i] = 0.f;
  }

  // issue one TDM weight-tile DMA into B buffer `buf` (wave 0 only)
  auto issueB = [&](int buf, int tap, int c0) {
    if (wid == 0) {
      unsigned long long ga = (unsigned long long)wpk
          + 4ull * ((unsigned long long)(tap * KPp + (c0 >> 1)) * (unsigned)Cout
                    + (unsigned)N0);
      u32x4 g0;
      g0[0] = 1u;                                        // count=1, user descriptor
      g0[1] = (unsigned)(2 * ABYTES + buf * BBYTES);     // lds_addr
      g0[2] = (unsigned)ga;                              // global_addr[31:0]
      g0[3] = (unsigned)((ga >> 32) & 0x1FFFFFFu) | (2u << 30);  // addr hi | type=2
      int dim0 = Cout - N0;                              // valid dwords from tile start
      i32x8 g1;
      g1[0] = (2 << 16) | (1 << 20) | (5 << 22) | (3 << 25); // 4B elems, pad 64dw->+4dw
      g1[1] = (dim0 & 0xFFFF) << 16;                     // tensor_dim0 lo16
      g1[2] = ((dim0 >> 16) & 0xFFFF) | (0x7FFF << 16);  // dim0 hi | dim1 lo (large)
      g1[3] = (64 << 16);                                // dim1 hi=0 | tile_dim0=64
      g1[4] = 16;                                        // tile_dim1=16, tile_dim2=0
      g1[5] = Cout;                                      // tensor_dim0_stride lo32
      g1[6] = 0;
      g1[7] = 0;
      i32x4 g2; g2[0] = 0; g2[1] = 0; g2[2] = 0; g2[3] = 0;
      i32x4 g3 = g2;
      asm volatile("tensor_load_to_lds %0, %1, %2, %3"
                   :: "s"(g0), "s"(g1), "s"(g2), "s"(g3) : "memory");
    }
  };

  // stage A tile (128 rows x 16 packed f16-pairs) into buffer `buf`, branch-free
  auto stageA = [&](int buf, int oy, int ox, int c0) {
    unsigned* Ab = As32 + buf * (128 * APW);
    for (int i = tid; i < 128 * 16; i += TB) {
      int mloc = i & 127, uc = i >> 7;
      int m   = M0 + mloc;
      int inm = m < M;
      int mm  = inm ? m : 0;
      int ho  = mm >> lwo, wo = mm & Wm1;
      int hi  = ho * stride + oy;
      int wi  = wo * stride + ox;
      int ins = inm & (hi >= 0) & (hi < H) & (wi >= 0) & (wi < W);
      int hic = ins ? hi : 0;
      int wic = ins ? wi : 0;
      int ci  = c0 + 2 * uc;
      int cia = ci     < Cin ? ci     : Cin - 1;
      int cib = ci + 1 < Cin ? ci + 1 : Cin - 1;
      float v0 = x[(((size_t)b * Cx + ci0 + cia) * H + hic) * W + wic];
      float v1 = x[(((size_t)b * Cx + ci0 + cib) * H + hic) * W + wic];
      F16x2 p;
      p.h[0] = (_Float16)((ins & (ci     < Cin)) ? v0 : 0.f);
      p.h[1] = (_Float16)((ins & (ci + 1 < Cin)) ? v1 : 0.f);
      Ab[mloc * APW + uc] = p.u;
    }
  };

  const int nchunks = (Cin + 31) >> 5;
  const int nsteps  = 9 * nchunks;

  // ---- prologue: fill buffer 0 with step 0 (tap 0, chunk 0) ----
  issueB(0, 0, 0);
  stageA(0, -dil, -dil, 0);
  if (wid == 0) __builtin_amdgcn_s_wait_tensorcnt(0);
  __syncthreads();

  int tap = 0, c0 = 0;
  for (int step = 0; step < nsteps; ++step) {
    const int buf = step & 1;
    // next step coordinates (incremental; no runtime division)
    int ntap = tap, nc0 = c0 + 32;
    if (nc0 >= Cin) { nc0 = 0; ntap = tap + 1; }
    const bool hasNext = (step + 1 < nsteps);

    if (hasNext) issueB(buf ^ 1, ntap, nc0);   // DMA overlaps WMMA + A staging

    // ---- fragments from current buffer: 32-bit LDS reads -> v16h ----
    const unsigned* Ab = As32 + buf * (128 * APW);
    const unsigned* Bb = Bs32 + buf * (16 * BPW);
    const int am0 = wm * 32 + l16;
    const int am1 = am0 + 16;
    v8u au0, au1, bu0, bu1;
#pragma unroll
    for (int v = 0; v < 8; ++v) {
      int kc = hl * 8 + v;                        // uint col: K pair index
      au0[v] = Ab[am0 * APW + kc];
      au1[v] = Ab[am1 * APW + kc];
    }
#pragma unroll
    for (int v = 0; v < 8; ++v) {
      int kp = hl * 8 + v;                        // K pair row
      bu0[v] = Bb[kp * BPW + wn * 32 + l16];
      bu1[v] = Bb[kp * BPW + wn * 32 + 16 + l16];
    }
    v16h a0  = __builtin_bit_cast(v16h, au0);
    v16h a1  = __builtin_bit_cast(v16h, au1);
    v16h bf0 = __builtin_bit_cast(v16h, bu0);
    v16h bf1 = __builtin_bit_cast(v16h, bu1);
    acc[0][0] = __builtin_amdgcn_wmma_f32_16x16x32_f16(false, a0, false, bf0,
                                                       (short)0, acc[0][0], false, false);
    acc[0][1] = __builtin_amdgcn_wmma_f32_16x16x32_f16(false, a0, false, bf1,
                                                       (short)0, acc[0][1], false, false);
    acc[1][0] = __builtin_amdgcn_wmma_f32_16x16x32_f16(false, a1, false, bf0,
                                                       (short)0, acc[1][0], false, false);
    acc[1][1] = __builtin_amdgcn_wmma_f32_16x16x32_f16(false, a1, false, bf1,
                                                       (short)0, acc[1][1], false, false);

    if (hasNext) {
      int nky = ntap / 3, nkx = ntap - 3 * nky;   // ntap < 9, tiny constants
      stageA(buf ^ 1, (nky - 1) * dil, (nkx - 1) * dil, nc0);
      if (wid == 0) __builtin_amdgcn_s_wait_tensorcnt(0);
    }
    __syncthreads();
    tap = ntap; c0 = nc0;
  }

  // ---- epilogue: bias + LeakyReLU; C layout: VGPR r -> M = hl*8+r, N = l16 ----
  const int n0 = N0 + wn * 32 + l16;
  const int n1 = n0 + 16;
  const float bz0 = bias[n0 < Cout ? n0 : Cout - 1];
  const float bz1 = bias[n1 < Cout ? n1 : Cout - 1];
  const size_t nstr = (size_t)M;   // channel stride in output plane
#pragma unroll
  for (int R = 0; R < 2; ++R)
#pragma unroll
  for (int r = 0; r < 8; ++r) {
    int m = M0 + wm * 32 + R * 16 + hl * 8 + r;
    if (m < M) {
      int ho = m >> lwo, wo = m & Wm1;
      size_t base = (((size_t)b * Cy + co0) * Ho + ho) * Wo + wo;
      if (n0 < Cout) {
        float v = acc[R][0][r] + bz0;
        if (relu) v = leakyf(v);
        y[base + (size_t)n0 * nstr] = v;
      }
      if (n1 < Cout) {
        float v = acc[R][1][r] + bz1;
        if (relu) v = leakyf(v);
        y[base + (size_t)n1 * nstr] = v;
      }
    }
  }
}

// ---------------------------------------------------------------------------
// Elementwise helpers
// ---------------------------------------------------------------------------
__global__ void zero_k(float* p, long n) {
  long i = (long)blockIdx.x * blockDim.x + threadIdx.x;
  if (i < n) p[i] = 0.f;
}

__global__ void add_k(float* d, const float* s, long n) {
  long i = (long)blockIdx.x * blockDim.x + threadIdx.x;
  if (i < n) d[i] += s[i];
}

__global__ void zero_ch_k(float* dst, int Cd, int off, int C, int H, int W, long n) {
  long i = (long)blockIdx.x * blockDim.x + threadIdx.x;
  if (i >= n) return;
  int w = (int)(i % W); long t = i / W;
  int h = (int)(t % H); t /= H;
  int c = (int)(t % C); int b = (int)(t / C);
  dst[(((long)b * Cd + off + c) * H + h) * W + w] = 0.f;
}

__global__ void copy_ch_k(const float* __restrict__ src, int Cs, int so,
                          float* __restrict__ dst, int Cd, int dof,
                          int C, int H, int W, long n) {
  long i = (long)blockIdx.x * blockDim.x + threadIdx.x;
  if (i >= n) return;
  int w = (int)(i % W); long t = i / W;
  int h = (int)(t % H); t /= H;
  int c = (int)(t % C); int b = (int)(t / C);
  dst[(((long)b * Cd + dof + c) * H + h) * W + w] =
      src[(((long)b * Cs + so + c) * H + h) * W + w];
}

// bilinear resize, align_corners=True, optional output gain
__global__ void resize_k(const float* __restrict__ src, float* __restrict__ dst,
                         int C, int H, int W, int H2, int W2, float gain, int B) {
  long n = (long)B * C * H2 * W2;
  long i = (long)blockIdx.x * blockDim.x + threadIdx.x;
  if (i >= n) return;
  int w = (int)(i % W2); long t = i / W2;
  int h = (int)(t % H2); t /= H2;
  int c = (int)(t % C);  int b = (int)(t / C);
  float ys = (H2 > 1) ? h * (float)(H - 1) / (float)(H2 - 1) : 0.f;
  float xs = (W2 > 1) ? w * (float)(W - 1) / (float)(W2 - 1) : 0.f;
  int y0 = (int)floorf(ys), x0 = (int)floorf(xs);
  float wy = ys - y0, wx = xs - x0;
  int y1 = y0 + 1 < H ? y0 + 1 : H - 1;
  int x1 = x0 + 1 < W ? x0 + 1 : W - 1;
  const float* p = src + ((long)b * C + c) * H * W;
  float v00 = p[(long)y0 * W + x0], v01 = p[(long)y0 * W + x1];
  float v10 = p[(long)y1 * W + x0], v11 = p[(long)y1 * W + x1];
  dst[i] = gain * ((v00 * (1.f - wx) + v01 * wx) * (1.f - wy)
                 + (v10 * (1.f - wx) + v11 * wx) * wy);
}

// grid_sample(bilinear, zeros, align_corners=True) + validity mask
__global__ void warp_k(const float* __restrict__ x, const float* __restrict__ flow,
                       float* __restrict__ out, int C, int H, int W,
                       int h_im, int w_im, int B) {
  long n = (long)B * C * H * W;
  long i = (long)blockIdx.x * blockDim.x + threadIdx.x;
  if (i >= n) return;
  int w = (int)(i % W); long t = i / W;
  int h = (int)(t % H); t /= H;
  int c = (int)(t % C); int b = (int)(t / C);
  float u = flow[(((long)b * 2 + 0) * H + h) * W + w] * 2.f / (float)(w_im - 1) / DIVF;
  float v = flow[(((long)b * 2 + 1) * H + h) * W + w] * 2.f / (float)(h_im - 1) / DIVF;
  float gx = (W > 1) ? -1.f + 2.f * w / (float)(W - 1) : -1.f;
  float gy = (H > 1) ? -1.f + 2.f * h / (float)(H - 1) : -1.f;
  float px = (gx + u + 1.f) * 0.5f * (W - 1);
  float py = (gy + v + 1.f) * 0.5f * (H - 1);
  float x0f = floorf(px), y0f = floorf(py);
  float fx = px - x0f, fy = py - y0f;
  int x0 = (int)x0f, y0 = (int)y0f;
  const float* p = x + ((long)b * C + c) * H * W;
  float acc = 0.f, wsum = 0.f;
#pragma unroll
  for (int dy = 0; dy < 2; ++dy)
#pragma unroll
  for (int dx = 0; dx < 2; ++dx) {
    int xi = x0 + dx, yi = y0 + dy;
    float wg = (dx ? fx : 1.f - fx) * (dy ? fy : 1.f - fy);
    int inb = (xi >= 0) & (xi < W) & (yi >= 0) & (yi < H);
    int xc = xi < 0 ? 0 : (xi > W - 1 ? W - 1 : xi);
    int yc = yi < 0 ? 0 : (yi > H - 1 ? H - 1 : yi);
    float g = p[(long)yc * W + xc];
    acc  += inb ? g * wg : 0.f;
    wsum += inb ? wg : 0.f;
  }
  out[i] = (wsum >= 1.f) ? acc : 0.f;
}

// FlowNet cost volume (search 4 -> 81 channels), mean over C, leaky fused
__global__ void corr_k(const float* __restrict__ a, const float* __restrict__ bw,
                       float* __restrict__ feats, int C, int H, int W,
                       int Cf, int off, int B) {
  long n = (long)B * 81 * H * W;
  long i = (long)blockIdx.x * blockDim.x + threadIdx.x;
  if (i >= n) return;
  int w = (int)(i % W); long t = i / W;
  int h = (int)(t % H); t /= H;
  int d = (int)(t % 81); int b = (int)(t / 81);
  int dy = d / 9 - 4, dx = d % 9 - 4;
  int h2 = h + dy, w2 = w + dx;
  float s = 0.f;
  if (h2 >= 0 && h2 < H && w2 >= 0 && w2 < W) {
    const float* pa = a  + ((long)b * C) * H * W + (long)h  * W + w;
    const float* pb = bw + ((long)b * C) * H * W + (long)h2 * W + w2;
    long cs = (long)H * W;
    for (int c = 0; c < C; ++c) s += pa[c * cs] * pb[c * cs];
  }
  s /= (float)C;
  feats[(((long)b * Cf + off + d) * H + h) * W + w] = leakyf(s);
}

// ---------------------------------------------------------------------------
struct CP { const float* w; const float* b; };

extern "C" void kernel_launch(void* const* d_in, const int* in_sizes, int n_in,
                              void* d_out, int out_size, void* d_ws, size_t ws_size,
                              hipStream_t stream)
{
  (void)in_sizes; (void)n_in; (void)out_size; (void)ws_size;
  const int B = 4, HIM = 256, WIM = 512;
  const float* x1 = (const float*)d_in[0];
  const float* x2 = (const float*)d_in[1];

  // params flattened in setup_inputs() insertion order: fpe, est, ctx
  int cur = 2;
  CP fpe[6][2];
  for (int l = 0; l < 6; ++l)
    for (int j = 0; j < 2; ++j) {
      fpe[l][j].w = (const float*)d_in[cur]; fpe[l][j].b = (const float*)d_in[cur + 1]; cur += 2;
    }
  CP est[5][7];
  for (int l = 0; l < 5; ++l)
    for (int j = 0; j < 7; ++j) {
      est[l][j].w = (const float*)d_in[cur]; est[l][j].b = (const float*)d_in[cur + 1]; cur += 2;
    }
  CP ctx[8];  // 6 dilated convs + flow + occ heads
  for (int j = 0; j < 8; ++j) {
    ctx[j].w = (const float*)d_in[cur]; ctx[j].b = (const float*)d_in[cur + 1]; cur += 2;
  }

  // bump allocator over workspace (deterministic per call)
  char* wbase = (char*)d_ws;
  size_t woff = 0;
  auto alloc = [&](size_t elems) -> float* {
    float* p = (float*)(wbase + woff);
    woff += (elems * sizeof(float) + 255) & ~(size_t)255;
    return p;
  };
  auto nb = [](size_t n) { return dim3((unsigned)((n + TB - 1) / TB)); };

  // pack weights (OIHW f32 -> TDM-friendly f16 pairs), then implicit-GEMM conv
  auto conv = [&](const float* xin, int Cx, int ci0, int Cin, int H, int W, CP p,
                  float* yout, int Cy, int co0, int Cout, int Ho, int Wo,
                  int stride, int dil, int relu) {
    int KPp = 16 * ((Cin + 31) / 32);
    size_t pkN = (size_t)9 * KPp * Cout;
    unsigned* wpk = (unsigned*)alloc(pkN);
    pack_w_k<<<nb(pkN), TB, 0, stream>>>(p.w, wpk, Cin, Cout, KPp, (long)pkN);
    int lwo = __builtin_ctz((unsigned)Wo);
    dim3 g((unsigned)((Ho * Wo + 127) / 128), (unsigned)((Cout + 63) / 64), (unsigned)B);
    conv3x3_wmma<<<g, TB, SMEMB, stream>>>(xin, Cx, ci0, Cin, H, W, wpk, KPp, p.b,
                                           yout, Cy, co0, Cout, Ho, Wo, lwo,
                                           stride, dil, relu);
  };

  // ---- feature pyramids for both images ----
  const int NCH[7] = {3, 16, 32, 64, 96, 128, 196};
  int PH[6], PW[6];
  { int h = HIM, w = WIM; for (int l = 0; l < 6; ++l) { h >>= 1; w >>= 1; PH[l] = h; PW[l] = w; } }

  float* pyr[2][6];
  for (int im = 0; im < 2; ++im) {
    const float* cf = im ? x2 : x1;
    int h = HIM, w = WIM, cc = 3;
    for (int l = 0; l < 6; ++l) {
      int h2 = PH[l], w2 = PW[l], co = NCH[l + 1];
      float* t = alloc((size_t)B * co * h2 * w2);
      conv(cf, cc, 0, cc, h, w, fpe[l][0], t, co, 0, co, h2, w2, 2, 1, 1);
      float* f = alloc((size_t)B * co * h2 * w2);
      conv(t, co, 0, co, h2, w2, fpe[l][1], f, co, 0, co, h2, w2, 1, 1, 1);
      pyr[im][l] = f; cf = f; cc = co; h = h2; w = w2;
    }
  }

  // ---- coarse-to-fine estimation (reference level l uses pyramid level 5-l) ----
  float *flow = nullptr, *occ = nullptr, *flow_b = nullptr, *occ_b = nullptr;
  int pH = 0, pW = 0;
  for (int l = 0; l <= 4; ++l) {
    int lev = 5 - l;
    int ch = NCH[lev + 1];
    int H = PH[lev], W = PW[lev];
    float* f1 = pyr[0][lev];
    float* f2 = pyr[1][lev];
    size_t hw = (size_t)H * W;

    if (l == 0) {
      flow = alloc((size_t)B * 2 * hw); occ = alloc((size_t)B * hw);
      flow_b = alloc((size_t)B * 2 * hw); occ_b = alloc((size_t)B * hw);
      zero_k<<<nb((size_t)B * 2 * hw), TB, 0, stream>>>(flow, (long)(B * 2 * hw));
      zero_k<<<nb((size_t)B * hw), TB, 0, stream>>>(occ, (long)(B * hw));
      zero_k<<<nb((size_t)B * 2 * hw), TB, 0, stream>>>(flow_b, (long)(B * 2 * hw));
      zero_k<<<nb((size_t)B * hw), TB, 0, stream>>>(occ_b, (long)(B * hw));
    } else {
      float* nf = alloc((size_t)B * 2 * hw);
      resize_k<<<nb((size_t)B * 2 * hw), TB, 0, stream>>>(flow, nf, 2, pH, pW, H, W, 1.f, B);
      flow = nf;
      float* no = alloc((size_t)B * hw);
      resize_k<<<nb((size_t)B * hw), TB, 0, stream>>>(occ, no, 1, pH, pW, H, W, 1.f, B);
      occ = no;
      float* nfb = alloc((size_t)B * 2 * hw);
      resize_k<<<nb((size_t)B * 2 * hw), TB, 0, stream>>>(flow_b, nfb, 2, pH, pW, H, W, 1.f, B);
      flow_b = nfb;
      float* nob = alloc((size_t)B * hw);
      resize_k<<<nb((size_t)B * hw), TB, 0, stream>>>(occ_b, nob, 1, pH, pW, H, W, 1.f, B);
      occ_b = nob;
    }

    const float* x2w = f2;
    const float* x1w = f1;
    if (l > 0) {
      float* t2 = alloc((size_t)B * ch * hw);
      warp_k<<<nb((size_t)B * ch * hw), TB, 0, stream>>>(f2, flow, t2, ch, H, W, HIM, WIM, B);
      x2w = t2;
      if (l < 4) {  // backward branch is dead at the final level
        float* t1 = alloc((size_t)B * ch * hw);
        warp_k<<<nb((size_t)B * ch * hw), TB, 0, stream>>>(f1, flow_b, t1, ch, H, W, HIM, WIM, B);
        x1w = t1;
      }
    }

    int ci = (l == 0) ? 83 : 83 + ch + 3;     // estimator input channels
    int Cf = 448 + ci + 3;                    // container: grows to x5 (+ctx tail)

    // build input stack [zeros2 | corr | feat | flow | occ] at channel 448
    auto build = [&](const float* fa, const float* fbw,
                     const float* fl, const float* oc) -> float* {
      float* F = alloc((size_t)B * Cf * hw);
      zero_ch_k<<<nb((size_t)B * 2 * hw), TB, 0, stream>>>(F, Cf, 448, 2, H, W,
                                                           (long)((size_t)B * 2 * hw));
      corr_k<<<nb((size_t)B * 81 * hw), TB, 0, stream>>>(fa, fbw, F, ch, H, W, Cf, 450, B);
      if (l > 0) {
        copy_ch_k<<<nb((size_t)B * ch * hw), TB, 0, stream>>>(fa, ch, 0, F, Cf, 448 + 83,
                                                              ch, H, W, (long)((size_t)B * ch * hw));
        copy_ch_k<<<nb((size_t)B * 2 * hw), TB, 0, stream>>>(fl, 2, 0, F, Cf, 448 + 83 + ch,
                                                             2, H, W, (long)((size_t)B * 2 * hw));
        copy_ch_k<<<nb((size_t)B * hw), TB, 0, stream>>>(oc, 1, 0, F, Cf, 448 + 85 + ch,
                                                         1, H, W, (long)((size_t)B * hw));
      }
      return F;
    };

    float* Ff = build(f1, x2w, flow, occ);
    float* Fb = (l < 4) ? build(f2, x1w, flow_b, occ_b) : nullptr;

    // dense estimator: concat-free by writing outputs at descending offsets
    auto dense = [&](float* F, int li, float** fo, float** oo) {
      CP* E = est[li];
      conv(F, Cf, 448, ci,       H, W, E[0], F, Cf, 320, 128, H, W, 1, 1, 1);
      conv(F, Cf, 320, 128 + ci, H, W, E[1], F, Cf, 192, 128, H, W, 1, 1, 1);
      conv(F, Cf, 192, 256 + ci, H, W, E[2], F, Cf,  96,  96, H, W, 1, 1, 1);
      conv(F, Cf,  96, 352 + ci, H, W, E[3], F, Cf,  32,  64, H, W, 1, 1, 1);
      conv(F, Cf,  32, 416 + ci, H, W, E[4], F, Cf,   0,  32, H, W, 1, 1, 1);
      float* f_ = alloc((size_t)B * 2 * hw);
      conv(F, Cf, 0, 448 + ci, H, W, E[5], f_, 2, 0, 2, H, W, 1, 1, 0);
      float* o_ = alloc((size_t)B * hw);
      conv(F, Cf, 0, 448 + ci, H, W, E[6], o_, 1, 0, 1, H, W, 1, 1, 0);
      *fo = f_; *oo = o_;
    };

    dense(Ff, l, &flow, &occ);
    if (l < 4) dense(Fb, l, &flow_b, &occ_b);

    if (l == 4) {  // context network on [x5 | flow | occ]
      copy_ch_k<<<nb((size_t)B * 2 * hw), TB, 0, stream>>>(flow, 2, 0, Ff, Cf, 448 + ci,
                                                           2, H, W, (long)((size_t)B * 2 * hw));
      copy_ch_k<<<nb((size_t)B * hw), TB, 0, stream>>>(occ, 1, 0, Ff, Cf, 450 + ci,
                                                       1, H, W, (long)((size_t)B * hw));
      float* cA = alloc((size_t)B * 128 * hw);
      float* cB = alloc((size_t)B * 128 * hw);
      conv(Ff, Cf, 0, Cf, H, W, ctx[0], cA, 128, 0, 128, H, W, 1, 1, 1);
      conv(cA, 128, 0, 128, H, W, ctx[1], cB, 128, 0, 128, H, W, 1, 2, 1);
      conv(cB, 128, 0, 128, H, W, ctx[2], cA, 128, 0, 128, H, W, 1, 4, 1);
      conv(cA, 128, 0, 128, H, W, ctx[3], cB,  96, 0,  96, H, W, 1, 8, 1);
      conv(cB,  96, 0,  96, H, W, ctx[4], cA,  64, 0,  64, H, W, 1, 16, 1);
      conv(cA,  64, 0,  64, H, W, ctx[5], cB,  32, 0,  32, H, W, 1, 1, 1);
      float* ff = alloc((size_t)B * 2 * hw);
      conv(cB, 32, 0, 32, H, W, ctx[6], ff, 2, 0, 2, H, W, 1, 1, 0);
      float* of = alloc((size_t)B * hw);
      conv(cB, 32, 0, 32, H, W, ctx[7], of, 1, 0, 1, H, W, 1, 1, 0);
      add_k<<<nb((size_t)B * 2 * hw), TB, 0, stream>>>(flow, ff, (long)((size_t)B * 2 * hw));
      add_k<<<nb((size_t)B * hw), TB, 0, stream>>>(occ, of, (long)((size_t)B * hw));
    }
    pH = H; pW = W;
  }

  // ---- upsample to full resolution; flow scaled by 1/DIV_FLOW ----
  float* out = (float*)d_out;
  size_t flowN = (size_t)B * 2 * HIM * WIM;
  resize_k<<<nb(flowN), TB, 0, stream>>>(flow, out, 2, pH, pW, HIM, WIM, 1.f / DIVF, B);
  size_t occN = (size_t)B * HIM * WIM;
  resize_k<<<nb(occN), TB, 0, stream>>>(occ, out + flowN, 1, pH, pW, HIM, WIM, 1.f, B);
}
